// SelfAttention_53395033424535
// MI455X (gfx1250) — matrix operand order repaired and checked by reading, not compiled, not run on hardware
//
#include <hip/hip_runtime.h>
#include <hip/hip_bf16.h>

typedef __bf16 bf16;
typedef __attribute__((ext_vector_type(16))) __bf16 v16bf;
typedef __attribute__((ext_vector_type(8)))  float  v8f;

union FragAB { v16bf v; uint4 q[2]; };
union V32    { uint4 q[2]; bf16 h[32]; };

// ---------------------------------------------------------------------------
// Async copy: 32 bytes (two b128) global -> LDS per lane, tracked by ASYNCcnt.
// IOFFSET is added to BOTH the LDS and global addresses (ISA 10.x async
// pseudocode), so one (lds, global) address pair covers both halves.
// LDS byte address = low 32 bits of the generic pointer (LDS aperture).
// ---------------------------------------------------------------------------
__device__ __forceinline__ void async_copy32(const bf16* lds_dst, const bf16* gsrc) {
    unsigned la = (unsigned)(unsigned long long)lds_dst;
    unsigned long long ga = (unsigned long long)gsrc;
    asm volatile("global_load_async_to_lds_b128 %0, %1, off\n\t"
                 "global_load_async_to_lds_b128 %0, %1, off offset:16"
                 :: "v"(la), "v"(ga) : "memory");
}
__device__ __forceinline__ void wait_async0() {
    asm volatile("s_wait_asynccnt 0x0" ::: "memory");
}

// ---------------------------------------------------------------------------
// Fragment loaders (CDNA5 WMMA 16-bit layouts, wave32)
// A-matrix 16x32 (MxK): lane&15 = M, lane>>4 selects K-half groups.
//   VGPR0-3: K = kh*8 + [0..7]   (16B)   VGPR4-7: K = 16 + kh*8 + [0..7]
// B-matrix 32x16 (KxN), fed from a transposed tile Bt[n][k]:
//   lane&15 = N, same K order as A: K = kh*8 + [0..7], then 16 + kh*8 + [0..7].
// ---------------------------------------------------------------------------
__device__ __forceinline__ v16bf lds_fragA(const bf16* base, int stride, int lane) {
    int m  = lane & 15;
    int kh = lane >> 4;
    FragAB f;
    const bf16* p = base + m * stride + kh * 8;
    f.q[0] = *(const uint4*)(p);
    f.q[1] = *(const uint4*)(p + 16);
    return f.v;
}

__device__ __forceinline__ v16bf lds_fragB(const bf16* base, int stride, int lane) {
    int n  = lane & 15;
    int kh = lane >> 4;
    FragAB f;
    const bf16* p = base + n * stride + kh * 8;
    f.q[0] = *(const uint4*)(p);
    f.q[1] = *(const uint4*)(p + 16);
    return f.v;
}

__device__ __forceinline__ v8f wmma_bf16(v16bf a, v16bf b, v8f c) {
    return __builtin_amdgcn_wmma_f32_16x16x32_bf16(
        false, a, false, b, (short)0, c, false, false);
}

// ---------------------------------------------------------------------------
// Elementwise converts
// ---------------------------------------------------------------------------
__global__ __launch_bounds__(256) void cvt_f32_bf16_kernel(
    const float* __restrict__ in, bf16* __restrict__ out, long n) {
    long i = (long)blockIdx.x * 256 + threadIdx.x;
    if (i < n) out[i] = (bf16)in[i];
}

// Wt[e][d] = W[d][e], fp32 -> bf16
__global__ __launch_bounds__(256) void cvt_transpose_kernel(
    const float* __restrict__ w, bf16* __restrict__ wt, int Dd) {
    long i = (long)blockIdx.x * 256 + threadIdx.x;   // i = e*D + d
    int e = (int)(i / Dd);
    int d = (int)(i % Dd);
    wt[i] = (bf16)w[(long)d * Dd + e];
}

// ---------------------------------------------------------------------------
// RoPE in place on bf16 (B,L,D) buffer, hd = 128, pairs (2i, 2i+1) per head
// ---------------------------------------------------------------------------
__global__ __launch_bounds__(256) void rope_kernel(
    bf16* __restrict__ buf, int Dd, int L) {
    long p = (long)blockIdx.x * 256 + threadIdx.x;   // pair index
    int halfD = Dd >> 1;
    long row = p / halfD;
    int  cp  = (int)(p % halfD);
    int  i   = cp & 63;                // within head, hd/2 = 64
    int  h   = cp >> 6;
    int  col = h * 128 + 2 * i;
    int  pos = (int)(row % L);
    float freq = __powf(10000.0f, -(float)(2 * i) / 128.0f);
    float s, c;
    __sincosf((float)pos * freq, &s, &c);
    long base = row * Dd + col;
    float x1 = (float)buf[base];
    float x2 = (float)buf[base + 1];
    buf[base]     = (bf16)(x1 * c - x2 * s);
    buf[base + 1] = (bf16)(x2 * c + x1 * s);
}

// ---------------------------------------------------------------------------
// bf16 GEMM: C[M,N] = A[M,K] * Bt[N,K]^T + bias[N]
// 128x128 tile per 256-thread (8-wave) workgroup; wave = 2x4 fragments.
// Double-buffered LDS, async global->LDS staging overlapped with WMMA.
// ---------------------------------------------------------------------------
template <bool F32OUT>
__global__ __launch_bounds__(256) void gemm_bf16_kernel(
    const bf16* __restrict__ A, const bf16* __restrict__ Bt,
    const float* __restrict__ bias, void* __restrict__ Cout,
    int M, int N, int K) {
    __shared__ __align__(16) bf16 lA[2][128 * 40];
    __shared__ __align__(16) bf16 lB[2][128 * 40];

    int tid  = threadIdx.x;
    int lane = tid & 31;
    int w    = tid >> 5;
    int waveM = w & 3;        // 4 waves over M (32 rows each)
    int waveN = w >> 2;       // 2 waves over N (64 cols each)
    int m0 = blockIdx.x * 128;
    int n0 = blockIdx.y * 128;

    int lrow = tid >> 1;      // 128 rows, 2 threads per row
    int lseg = tid & 1;       // 16-half segment

    v8f acc[2][4] = {};

    auto stage_tile = [&](int buf, int k0) {
        async_copy32(&lA[buf][lrow * 40 + lseg * 16],
                     A + (long)(m0 + lrow) * K + k0 + lseg * 16);
        async_copy32(&lB[buf][lrow * 40 + lseg * 16],
                     Bt + (long)(n0 + lrow) * K + k0 + lseg * 16);
    };

    stage_tile(0, 0);
    wait_async0();
    __syncthreads();

    int nsteps = K / 32;
    for (int s = 0; s < nsteps; ++s) {
        int cur = s & 1;
        if (s + 1 < nsteps) stage_tile(cur ^ 1, (s + 1) * 32);

        const bf16* curA = lA[cur];
        const bf16* curB = lB[cur];
        v16bf af[2], bfr[4];
#pragma unroll
        for (int mf = 0; mf < 2; ++mf)
            af[mf] = lds_fragA(curA + (waveM * 32 + mf * 16) * 40, 40, lane);
#pragma unroll
        for (int nf = 0; nf < 4; ++nf)
            bfr[nf] = lds_fragB(curB + (waveN * 64 + nf * 16) * 40, 40, lane);
#pragma unroll
        for (int mf = 0; mf < 2; ++mf)
#pragma unroll
            for (int nf = 0; nf < 4; ++nf)
                acc[mf][nf] = wmma_bf16(af[mf], bfr[nf], acc[mf][nf]);

        wait_async0();          // next tile landed in the other buffer
        __syncthreads();
    }

    // Epilogue: C layout lane&15 = N, lane>>4 picks M+8, VGPR r = M row
    int nn = lane & 15;
    int mh = lane >> 4;
#pragma unroll
    for (int nf = 0; nf < 4; ++nf) {
        int col = n0 + waveN * 64 + nf * 16 + nn;
        float bv = bias[col];
#pragma unroll
        for (int mf = 0; mf < 2; ++mf) {
#pragma unroll
            for (int r = 0; r < 8; ++r) {
                int row = m0 + waveM * 32 + mf * 16 + mh * 8 + r;
                float val = acc[mf][nf][r] + bv;
                if constexpr (F32OUT)
                    ((float*)Cout)[(long)row * N + col] = val;
                else
                    ((bf16*)Cout)[(long)row * N + col] = (bf16)val;
            }
        }
    }
}

// ---------------------------------------------------------------------------
// Flash attention, causal. Grid: (L/64, H, B). Block: 256 threads = 8 waves.
// 64 query rows per block, key chunks of 64, online softmax.
// Q tile and K chunks staged via async global->LDS.
// ---------------------------------------------------------------------------
__global__ __launch_bounds__(256) void attn_kernel(
    const bf16* __restrict__ Q, const bf16* __restrict__ K,
    const bf16* __restrict__ V, bf16* __restrict__ O) {
    constexpr int L = 256, D = 2048, HD = 128;
    __shared__ __align__(16) bf16 qt[64 * 136];   // Q tile   (stride 136)
    __shared__ __align__(16) bf16 kv[128 * 72];   // K tile (stride 136) / V^T tile (stride 72)
    __shared__ __align__(16) float sS[64 * 68];   // score chunk
    __shared__ __align__(16) bf16 sP[64 * 72];    // exp(S - m) chunk
    __shared__ float rowm[64], rowl[64], rowscale[64];

    int tid = threadIdx.x, lane = tid & 31, w = tid >> 5;
    int qb = blockIdx.x, h = blockIdx.y, b = blockIdx.z;
    int q0 = qb * 64;
    int waveM = w & 3;   // 16 query rows
    int waveN = w >> 2;  // 64 dims

    long baseQ = ((long)b * L + q0) * D + h * HD;
    int  srow = tid >> 2, sseg = tid & 3;

    // async-stage Q tile (waited inside first chunk iteration)
    async_copy32(&qt[srow * 136 + sseg * 32], Q + baseQ + (long)srow * D + sseg * 32);
    if (tid < 64) { rowm[tid] = -3.0e38f; rowl[tid] = 0.0f; }

    v8f o[4] = {};
    const float scale = 0.08838834764831845f;   // 1/sqrt(128)

    for (int c = 0; c <= qb; ++c) {
        __syncthreads();   // previous-iteration kv readers are done
        // async-stage K chunk (natural [key][dim] layout == Bt for Q*K^T)
        async_copy32(&kv[srow * 136 + sseg * 32],
                     K + ((long)b * L + c * 64 + srow) * D + h * HD + sseg * 32);
        wait_async0();
        __syncthreads();

        // S = Q * K^T (2 fragments per wave), masked + scaled into LDS
#pragma unroll
        for (int e = 0; e < 2; ++e) {
            int ni = waveN * 2 + e;
            v8f s8 = {};
#pragma unroll
            for (int kk = 0; kk < 4; ++kk) {
                v16bf a  = lds_fragA(qt + waveM * 16 * 136 + kk * 32, 136, lane);
                v16bf bb = lds_fragB(kv + ni * 16 * 136 + kk * 32, 136, lane);
                s8 = wmma_bf16(a, bb, s8);
            }
            int nn = lane & 15, mh = lane >> 4;
#pragma unroll
            for (int r = 0; r < 8; ++r) {
                int rl = waveM * 16 + mh * 8 + r;
                int kl = ni * 16 + nn;
                bool ok = (c * 64 + kl) <= (q0 + rl);
                sS[rl * 68 + kl] = ok ? s8[r] * scale : -3.0e38f;
            }
        }
        __syncthreads();

        {   // online softmax over this 64-key chunk: 4 lanes per row
            int row = tid >> 2, sub = tid & 3;
            float mc = -3.0e38f;
            for (int j = sub; j < 64; j += 4) mc = fmaxf(mc, sS[row * 68 + j]);
            mc = fmaxf(mc, __shfl_xor(mc, 1, 32));
            mc = fmaxf(mc, __shfl_xor(mc, 2, 32));
            float mold = rowm[row];
            float mnew = fmaxf(mold, mc);
            float ls = 0.0f;
            for (int j = sub; j < 64; j += 4) {
                float ev = __expf(sS[row * 68 + j] - mnew);
                sP[row * 72 + j] = (bf16)ev;
                ls += ev;
            }
            ls += __shfl_xor(ls, 1, 32);
            ls += __shfl_xor(ls, 2, 32);
            if (sub == 0) {
                float resc = __expf(mold - mnew);
                rowscale[row] = resc;
                rowm[row] = mnew;
                rowl[row] = rowl[row] * resc + ls;
            }
        }
        __syncthreads();

        {   // stage V chunk transposed: kv[dim*72 + key]
            int key = tid & 63, dseg = tid >> 6;
            long gv = ((long)b * L + c * 64 + key) * D + h * HD + dseg * 32;
            V32 vr;
            vr.q[0] = ((const uint4*)(V + gv))[0];
            vr.q[1] = ((const uint4*)(V + gv))[1];
#pragma unroll
            for (int i = 0; i < 32; ++i)
                kv[(dseg * 32 + i) * 72 + key] = vr.h[i];
        }
        __syncthreads();

        {   // rescale O and accumulate P*V
            int mh = lane >> 4;
            float sc[8];
#pragma unroll
            for (int r = 0; r < 8; ++r) sc[r] = rowscale[waveM * 16 + mh * 8 + r];
#pragma unroll
            for (int nf = 0; nf < 4; ++nf)
#pragma unroll
                for (int r = 0; r < 8; ++r) o[nf][r] *= sc[r];
#pragma unroll
            for (int kk = 0; kk < 2; ++kk) {
                v16bf a = lds_fragA(sP + waveM * 16 * 72 + kk * 32, 72, lane);
#pragma unroll
                for (int nf = 0; nf < 4; ++nf) {
                    v16bf bb = lds_fragB(kv + (waveN * 64 + nf * 16) * 72 + kk * 32, 72, lane);
                    o[nf] = wmma_bf16(a, bb, o[nf]);
                }
            }
        }
    }
    __syncthreads();

    {   // epilogue: O /= rowsum, bf16 store to (B,L,D)
        int nn = lane & 15, mh = lane >> 4;
        float inv[8];
#pragma unroll
        for (int r = 0; r < 8; ++r) inv[r] = 1.0f / rowl[waveM * 16 + mh * 8 + r];
#pragma unroll
        for (int nf = 0; nf < 4; ++nf)
#pragma unroll
            for (int r = 0; r < 8; ++r) {
                int rl  = waveM * 16 + mh * 8 + r;
                int col = waveN * 64 + nf * 16 + nn;
                O[((long)b * L + q0 + rl) * D + h * HD + col] = (bf16)(o[nf][r] * inv[r]);
            }
    }
}

// ---------------------------------------------------------------------------
// Host launcher
// ---------------------------------------------------------------------------
extern "C" void kernel_launch(void* const* d_in, const int* in_sizes, int n_in,
                              void* d_out, int out_size, void* d_ws, size_t ws_size,
                              hipStream_t stream) {
    constexpr int B = 64, L = 256, D = 2048, H = 16;
    constexpr long NX = (long)B * L * D;   // 33,554,432
    constexpr long NW = (long)D * D;       //  4,194,304
    constexpr int  M  = B * L;             // 16384

    const float* x  = (const float*)d_in[0];
    const float* wq = (const float*)d_in[1];
    const float* bq = (const float*)d_in[2];
    const float* wk = (const float*)d_in[3];
    const float* bk = (const float*)d_in[4];
    const float* wv = (const float*)d_in[5];
    const float* bv = (const float*)d_in[6];
    const float* wo = (const float*)d_in[7];
    const float* bo = (const float*)d_in[8];

    char* p = (char*)d_ws;
    bf16* xb  = (bf16*)p; p += NX * 2;
    bf16* wqt = (bf16*)p; p += NW * 2;
    bf16* wkt = (bf16*)p; p += NW * 2;
    bf16* wvt = (bf16*)p; p += NW * 2;
    bf16* wot = (bf16*)p; p += NW * 2;
    bf16* qb_ = (bf16*)p; p += NX * 2;
    bf16* kb_ = (bf16*)p; p += NX * 2;
    bf16* vb_ = (bf16*)p; p += NX * 2;
    bf16* ob_ = (bf16*)p; p += NX * 2;

    // 1) pack to bf16 (weights pre-transposed)
    cvt_f32_bf16_kernel<<<(int)(NX / 256), 256, 0, stream>>>(x, xb, NX);
    cvt_transpose_kernel<<<(int)(NW / 256), 256, 0, stream>>>(wq, wqt, D);
    cvt_transpose_kernel<<<(int)(NW / 256), 256, 0, stream>>>(wk, wkt, D);
    cvt_transpose_kernel<<<(int)(NW / 256), 256, 0, stream>>>(wv, wvt, D);
    cvt_transpose_kernel<<<(int)(NW / 256), 256, 0, stream>>>(wo, wot, D);

    // 2) Q/K/V projections (WMMA GEMMs, bf16 out)
    dim3 g(M / 128, D / 128);
    gemm_bf16_kernel<false><<<g, 256, 0, stream>>>(xb, wqt, bq, qb_, M, D, D);
    gemm_bf16_kernel<false><<<g, 256, 0, stream>>>(xb, wkt, bk, kb_, M, D, D);
    gemm_bf16_kernel<false><<<g, 256, 0, stream>>>(xb, wvt, bv, vb_, M, D, D);

    // 3) RoPE on Q and K (in place; buffers fully rewritten each call)
    rope_kernel<<<(int)(NX / 2 / 256), 256, 0, stream>>>(qb_, D, L);
    rope_kernel<<<(int)(NX / 2 / 256), 256, 0, stream>>>(kb_, D, L);

    // 4) causal flash attention (WMMA)
    attn_kernel<<<dim3(L / 64, H, B), 256, 0, stream>>>(qb_, kb_, vb_, ob_);

    // 5) output projection, f32 epilogue straight into d_out
    gemm_bf16_kernel<true><<<g, 256, 0, stream>>>(ob_, wot, bo, d_out, M, D, D);
}
